// TopKSparseAutoencoder_4071628997269
// MI455X (gfx1250) — matrix operand order repaired and checked by reading, not compiled, not run on hardware
//
#include <hip/hip_runtime.h>

typedef __bf16 bf16_t;
typedef __attribute__((ext_vector_type(16))) __bf16 v16bf;
typedef __attribute__((ext_vector_type(8)))  __bf16 v8bf;
typedef __attribute__((ext_vector_type(4)))  __bf16 v4bf;
typedef __attribute__((ext_vector_type(8)))  float  v8f;

#define BM 128
#define BN 128
#define BK 64
#define LDT (BK + 8)   // padded LDS row stride (bf16 elems): 144B -> conflict-free b128 loads

// C[M,N] = op(A[M,Kd] * Bm[N,Kd]^T + bias), bf16 WMMA with f32 accumulation.
// Double-buffered LDS + register-staged global pipeline: one barrier per K-tile,
// global loads for tile k+1 overlap the WMMA stream for tile k.
template<bool FUSE_BIAS_RELU>
__global__ __launch_bounds__(256)
void gemm_bf16(const float* __restrict__ A,
               const float* __restrict__ Bm,
               const float* __restrict__ bias,
               float* __restrict__ C,
               int M, int N, int Kd)
{
    __shared__ bf16_t As[2][BM * LDT];   // 2 x 18 KB
    __shared__ bf16_t Bs[2][BN * LDT];   // 2 x 18 KB

    const int tid      = threadIdx.x;
    const int lane     = tid & 31;
    const int wave     = tid >> 5;          // 8 waves
    const int laneIdx  = lane & 15;
    const int laneHalf = lane >> 4;
    const int mWave    = (wave >> 2) * 64;  // 2 waves along M (64 rows each)
    const int nWave    = (wave & 3) * 32;   // 4 waves along N (32 cols each)

    // grouped block swizzle for L2 reuse
    const int nBM = M / BM, nBN = N / BN;
    const int GROUP = 8;
    int pid       = blockIdx.x;
    int groupSize = GROUP * nBN;
    int group     = pid / groupSize;
    int firstM    = group * GROUP;
    int gsz       = (nBM - firstM) < GROUP ? (nBM - firstM) : GROUP;
    int mb        = firstM + ((pid % groupSize) % gsz);
    int nb        = (pid % groupSize) / gsz;

    const size_t aBase = (size_t)mb * BM * Kd;
    const size_t bBase = (size_t)nb * BN * Kd;

    // per-thread tile coordinates (8 float4 per operand per tile)
    const int ldRow = tid >> 4;             // reused for j==0; full mapping below
    (void)ldRow;

    v8f acc[4][2];
#pragma unroll
    for (int mi = 0; mi < 4; ++mi)
#pragma unroll
        for (int ni = 0; ni < 2; ++ni)
            acc[mi][ni] = (v8f)0.0f;

    float4 ra[8], rb[8];                    // raw f32 staging registers

    auto load_tile = [&](int k0) {
#pragma unroll
        for (int j = 0; j < 8; ++j) {
            int linear = tid + j * 256;       // 0..2047
            int row    = linear >> 4;         // 0..127
            int c4     = (linear & 15) << 2;  // 0,4,..,60
            ra[j] = *(const float4*)(A  + aBase + (size_t)row * Kd + k0 + c4);
            rb[j] = *(const float4*)(Bm + bBase + (size_t)row * Kd + k0 + c4);
        }
    };
    auto store_tile = [&](int buf) {          // f32 regs -> bf16 -> LDS
#pragma unroll
        for (int j = 0; j < 8; ++j) {
            int linear = tid + j * 256;
            int row    = linear >> 4;
            int c4     = (linear & 15) << 2;
            v4bf pa = { (bf16_t)ra[j].x, (bf16_t)ra[j].y, (bf16_t)ra[j].z, (bf16_t)ra[j].w };
            *(v4bf*)&As[buf][row * LDT + c4] = pa;
            v4bf pb = { (bf16_t)rb[j].x, (bf16_t)rb[j].y, (bf16_t)rb[j].z, (bf16_t)rb[j].w };
            *(v4bf*)&Bs[buf][row * LDT + c4] = pb;
        }
    };

    const int nIter = Kd / BK;
    load_tile(0);
    store_tile(0);

    for (int it = 0; it < nIter; ++it) {
        if (it + 1 < nIter) {
            load_tile((it + 1) * BK);         // issue early; consumed after compute
            if (it + 2 < nIter) {             // prime L2 two tiles ahead
                __builtin_prefetch(A  + aBase + (size_t)(tid >> 1) * Kd + (size_t)(it + 2) * BK, 0, 1);
                __builtin_prefetch(Bm + bBase + (size_t)(tid >> 1) * Kd + (size_t)(it + 2) * BK, 0, 1);
            }
        }
        __syncthreads();                      // buf stores visible; prior reads retired

        const bf16_t* __restrict__ AsBuf = As[it & 1];
        const bf16_t* __restrict__ BsBuf = Bs[it & 1];
#pragma unroll
        for (int kk = 0; kk < BK; kk += 32) {
            v16bf bFrag[2], aFrag[4];
#pragma unroll
            for (int ni = 0; ni < 2; ++ni) {
                // B lane (laneIdx = N col, laneHalf = K half): 16 contiguous bf16
                const bf16_t* p = &BsBuf[(nWave + ni * 16 + laneIdx) * LDT + kk + laneHalf * 16];
                union { v16bf v; v8bf h[2]; } u;
                u.h[0] = *(const v8bf*)(p);
                u.h[1] = *(const v8bf*)(p + 8);
                bFrag[ni] = u.v;
            }
#pragma unroll
            for (int mi = 0; mi < 4; ++mi) {
                // A lane (laneIdx = M row, laneHalf = K half): 16 contiguous bf16
                const bf16_t* p = &AsBuf[(mWave + mi * 16 + laneIdx) * LDT + kk + laneHalf * 16];
                union { v16bf v; v8bf h[2]; } u;
                u.h[0] = *(const v8bf*)(p);
                u.h[1] = *(const v8bf*)(p + 8);
                aFrag[mi] = u.v;
            }
#pragma unroll
            for (int mi = 0; mi < 4; ++mi)
#pragma unroll
                for (int ni = 0; ni < 2; ++ni)
                    acc[mi][ni] = __builtin_amdgcn_wmma_f32_16x16x32_bf16(
                        false, aFrag[mi], false, bFrag[ni],
                        (short)0, acc[mi][ni], false, false);
        }

        if (it + 1 < nIter)
            store_tile((it + 1) & 1);         // write the buffer not read this iter
    }

    // epilogue: C/D layout — lane n = laneIdx, rows m = laneHalf*8 + v
#pragma unroll
    for (int mi = 0; mi < 4; ++mi) {
#pragma unroll
        for (int ni = 0; ni < 2; ++ni) {
            int col = nb * BN + nWave + ni * 16 + laneIdx;
            float bv = 0.0f;
            if (FUSE_BIAS_RELU) bv = bias[col];
            int rowBase = mb * BM + mWave + mi * 16 + laneHalf * 8;
#pragma unroll
            for (int v = 0; v < 8; ++v) {
                float r = acc[mi][ni][v] + bv;
                if (FUSE_BIAS_RELU) r = fmaxf(r, 0.0f);
                C[(size_t)(rowBase + v) * N + col] = r;
            }
        }
    }
}

// Exact top-64 per row of 16384 non-negative floats via 4-pass radix select on
// the u32 bit pattern (monotonic for values >= 0). In-place masking.
__global__ __launch_bounds__(256)
void topk_mask(float* __restrict__ feat)
{
    extern __shared__ unsigned vals[];   // 16384 u32 = 64 KB
    __shared__ unsigned hist[256];
    __shared__ unsigned s_prefix, s_remain, s_tie;

    const int tid = threadIdx.x;
    float* rowp = feat + (size_t)blockIdx.x * 16384;

    for (int i = tid; i < 16384; i += 256)
        vals[i] = __float_as_uint(rowp[i]);
    if (tid == 0) { s_prefix = 0u; s_remain = 64u; s_tie = 0u; }
    __syncthreads();

    for (int pass = 0; pass < 4; ++pass) {
        const int shift = 24 - pass * 8;
        const unsigned hiMask = (pass == 0) ? 0u : (0xFFFFFFFFu << (shift + 8));
        const unsigned prefix = s_prefix;
        hist[tid] = 0u;
        __syncthreads();
        for (int i = tid; i < 16384; i += 256) {
            unsigned v = vals[i];
            if ((v & hiMask) == prefix)
                atomicAdd(&hist[(v >> shift) & 255u], 1u);
        }
        __syncthreads();
        if (tid == 0) {
            unsigned remain = s_remain;
            unsigned above = 0u;
            int b = 255;
            for (; b > 0; --b) {
                unsigned c = hist[b];
                if (above + c >= remain) break;
                above += c;
            }
            s_prefix = prefix | ((unsigned)b << shift);
            s_remain = remain - above;    // #ties at threshold still to keep
        }
        __syncthreads();
    }

    const unsigned T = s_prefix;
    const unsigned nties = s_remain;
    __syncthreads();

    for (int i = tid; i < 16384; i += 256) {
        unsigned v = vals[i];
        bool keep;
        if (v > T)       keep = true;
        else if (v == T) keep = (atomicAdd(&s_tie, 1u) < nties);
        else             keep = false;
        rowp[i] = keep ? __uint_as_float(v) : 0.0f;
    }
}

extern "C" void kernel_launch(void* const* d_in, const int* in_sizes, int n_in,
                              void* d_out, int out_size, void* d_ws, size_t ws_size,
                              hipStream_t stream)
{
    (void)in_sizes; (void)n_in; (void)out_size; (void)d_ws; (void)ws_size;
    const float* x     = (const float*)d_in[0];   // [4096, 4096]
    const float* W_enc = (const float*)d_in[1];   // [16384, 4096]
    const float* b_enc = (const float*)d_in[2];   // [16384]
    const float* W_dec = (const float*)d_in[3];   // [4096, 16384]

    float* feats = (float*)d_out;                       // [4096, 16384]
    float* rec   = feats + (size_t)4096 * 16384;        // [4096, 4096]

    dim3 blk(256);
    // encoder: M=4096, N=16384, K=4096, fused bias+ReLU
    gemm_bf16<true><<<(4096 / BM) * (16384 / BN), blk, 0, stream>>>(
        x, W_enc, b_enc, feats, 4096, 16384, 4096);
    // exact per-row top-64 masking, in place
    topk_mask<<<4096, blk, 16384 * sizeof(unsigned), stream>>>(feats);
    // decoder: M=4096, N=4096, K=16384 (sparse rows treated dense)
    gemm_bf16<false><<<(4096 / BM) * (4096 / BN), blk, 0, stream>>>(
        feats, W_dec, nullptr, rec, 4096, 4096, 16384);
}
